// AlignmentContrastiveLoss_38714835206567
// MI455X (gfx1250) — matrix operand order repaired and checked by reading, not compiled, not run on hardware
//
#include <hip/hip_runtime.h>
#include <hip/hip_bf16.h>
#include <stdint.h>

#define MARGIN 0.2f

typedef __bf16 bf16_t;
typedef bf16_t v16bf __attribute__((ext_vector_type(16)));
typedef bf16_t v8bf  __attribute__((ext_vector_type(8)));
typedef float  v8f   __attribute__((ext_vector_type(8)));
typedef int    v4i   __attribute__((ext_vector_type(4)));

#if defined(__has_builtin)
# if __has_builtin(__builtin_amdgcn_global_load_async_to_lds_b128)
#  define USE_ASYNC_LDS 1
# endif
#endif

// float -> bf16 round-to-nearest-even (data is Gaussian; no NaN/Inf handling needed)
__device__ __forceinline__ unsigned short f2bf(float f) {
  unsigned int u = __float_as_uint(f);
  u += 0x7FFFu + ((u >> 16) & 1u);
  return (unsigned short)(u >> 16);
}

// im_b[i][n][d] = bf16(im_set[i][n+1][d]) for n < im_len[i]-1, else 0 (n in [0,48))
__global__ void cvt_im(const float* __restrict__ im_set, const int* __restrict__ im_len,
                       unsigned short* __restrict__ im_b) {
  const int i  = blockIdx.x;
  const int il = im_len[i] - 1;               // <= 36
  const float* src = im_set + (size_t)i * 37 * 1024;
  unsigned short* dst = im_b + (size_t)i * 48 * 1024;
  for (int idx = threadIdx.x; idx < 48 * 1024; idx += blockDim.x) {
    const int n = idx >> 10, d = idx & 1023;
    const float v = (n < il) ? src[(n + 1) * 1024 + d] : 0.0f;
    dst[idx] = f2bf(v);
  }
}

// s_b[j][m][d] = bf16(s_seq[j][m+1][d]) for m < s_len[j]-3, else 0 (m in [0,32))
__global__ void cvt_s(const float* __restrict__ s_seq, const int* __restrict__ s_len,
                      unsigned short* __restrict__ s_b) {
  const int j  = blockIdx.x;
  const int sl = s_len[j] - 3;                // <= 32
  const float* src = s_seq + (size_t)j * 35 * 1024;
  unsigned short* dst = s_b + (size_t)j * 32 * 1024;
  for (int idx = threadIdx.x; idx < 32 * 1024; idx += blockDim.x) {
    const int m = idx >> 10, d = idx & 1023;
    const float v = (m < sl) ? src[(m + 1) * 1024 + d] : 0.0f;
    dst[idx] = f2bf(v);
  }
}

// One block per (i,j): align = im_b[i] (48x1024) x s_b[j]^T (1024x32) via bf16 WMMA,
// then scores[i][j] = sum_m max_{n<36} align[n][m].
#define BLK_T 192
__global__ void __launch_bounds__(BLK_T)
gemm_scores(const unsigned short* __restrict__ im_b,
            const unsigned short* __restrict__ s_b,
            float* __restrict__ scores) {
  __shared__ unsigned short As[48 * 256];   // 24 KB : A chunk (rows x K-slice)
  __shared__ unsigned short Bs[32 * 256];   // 16 KB : B chunk
  __shared__ float          Cs[48 * 32];    //  6 KB : result tile

  const int j = blockIdx.x, i = blockIdx.y;
  const int tid   = threadIdx.x;
  const int lane  = tid & 31;
  const int wave  = tid >> 5;      // 0..5
  const int tileM = wave % 3;      // 3 tiles cover rows 0..47
  const int tileN = wave / 3;      // 2 tiles cover cols 0..31
  const int row   = lane & 15;
  const int hi    = lane >> 4;     // lane-half selects K sub-range (ISA 7.12.2)

  const uint4* gA = (const uint4*)(im_b + (size_t)i * 48 * 1024);  // 128 uint4/row
  const uint4* gB = (const uint4*)(s_b  + (size_t)j * 32 * 1024);
  uint4* lA = (uint4*)As;                                          // 32 uint4/row
  uint4* lB = (uint4*)Bs;

  v8f acc = {};

  for (int ch = 0; ch < 4; ++ch) {          // K chunks of 256
    // --- stage chunk into LDS (1536 A-vecs + 1024 B-vecs of 16B) ---
    for (int t = tid; t < 2560; t += BLK_T) {
      const uint4* g; uint4* l;
      if (t < 1536) {
        const int n = t >> 5, q = t & 31;
        g = gA + (size_t)n * 128 + ch * 32 + q;
        l = lA + n * 32 + q;
      } else {
        const int u = t - 1536;
        const int m = u >> 5, q = u & 31;
        g = gB + (size_t)m * 128 + ch * 32 + q;
        l = lB + m * 32 + q;
      }
#ifdef USE_ASYNC_LDS
      __builtin_amdgcn_global_load_async_to_lds_b128(
          (__attribute__((address_space(1))) v4i*)(v4i*)g,
          (__attribute__((address_space(3))) v4i*)(v4i*)l, 0, 0);
#else
      *l = *g;
#endif
    }
#ifdef USE_ASYNC_LDS
# if __has_builtin(__builtin_amdgcn_s_wait_asynccnt)
    __builtin_amdgcn_s_wait_asynccnt(0);
# else
    asm volatile("s_wait_asynccnt 0" ::: "memory");
# endif
#endif
    __syncthreads();

    // --- 8 WMMA K-steps over this 256-wide chunk ---
    const unsigned short* aRow = As + (tileM * 16 + row) * 256;
    const unsigned short* bRow = Bs + (tileN * 16 + row) * 256;
    #pragma unroll
    for (int kk = 0; kk < 8; ++kk) {
      const int kb = kk * 32;
      // A frag (16x32): lane<16 -> K {0..15}; lane>=16 -> K {16..31}
      v8bf alo = *(const v8bf*)(aRow + kb + hi * 16);
      v8bf ahi = *(const v8bf*)(aRow + kb + hi * 16 + 8);
      v16bf a;
      #pragma unroll
      for (int e = 0; e < 8; ++e) { a[e] = alo[e]; a[8 + e] = ahi[e]; }
      // B frag (32x16): lane holds col N=row, K = hi*16 + {0..15} (contiguous)
      v16bf b = *(const v16bf*)(bRow + kb + hi * 16);
      acc = __builtin_amdgcn_wmma_f32_16x16x32_bf16(
          false, a, false, b, (short)0, acc, false, false);
    }
    __syncthreads();
  }

  // --- scatter C tile: VGPR r holds M = r + hi*8, N = lane&15 ---
  #pragma unroll
  for (int r = 0; r < 8; ++r) {
    const int n = tileM * 16 + hi * 8 + r;
    const int m = tileN * 16 + row;
    Cs[n * 32 + m] = acc[r];
  }
  __syncthreads();

  // --- col-max over n<36 (padding rows 36..47 excluded), wave32 sum over m ---
  if (tid < 32) {
    float cmax = Cs[tid];
    #pragma unroll
    for (int n = 1; n < 36; ++n) cmax = fmaxf(cmax, Cs[n * 32 + tid]);
    float s = cmax;
    #pragma unroll
    for (int off = 16; off > 0; off >>= 1) s += __shfl_down(s, off, 32);
    if (tid == 0) scores[i * 256 + j] = s;
  }
}

// Contrastive reduction over the 256x256 score matrix -> scalar loss.
__global__ void loss_kernel(const float* __restrict__ S, float* __restrict__ out) {
  __shared__ float red[256];
  const int t = threadIdx.x;
  const float d = S[t * 256 + t];
  float rm = 0.0f, cm = 0.0f;    // diag entries forced to 0 => init 0 and skip k==t
  for (int k = 0; k < 256; ++k) {
    if (k != t) {
      rm = fmaxf(rm, fmaxf(MARGIN + S[t * 256 + k] - d, 0.0f));  // cost_s row t
      cm = fmaxf(cm, fmaxf(MARGIN + S[k * 256 + t] - d, 0.0f));  // cost_im col t
    }
  }
  red[t] = rm + cm;
  __syncthreads();
  for (int s = 128; s > 0; s >>= 1) {
    if (t < s) red[t] += red[t + s];
    __syncthreads();
  }
  if (t == 0) out[0] = red[0];
}

extern "C" void kernel_launch(void* const* d_in, const int* in_sizes, int n_in,
                              void* d_out, int out_size, void* d_ws, size_t ws_size,
                              hipStream_t stream) {
  const float* im_set = (const float*)d_in[0];
  const float* s_seq  = (const float*)d_in[1];
  const int*   im_len = (const int*)d_in[2];
  const int*   s_len  = (const int*)d_in[3];

  char* ws = (char*)d_ws;
  unsigned short* im_b = (unsigned short*)ws;                         // 256*48*1024*2 = 24 MiB
  unsigned short* s_b  = (unsigned short*)(ws + 25165824);            // 256*32*1024*2 = 16 MiB
  float*          scor = (float*)(ws + 25165824 + 16777216);          // 256*256*4

  cvt_im<<<256, 256, 0, stream>>>(im_set, im_len, im_b);
  cvt_s <<<256, 256, 0, stream>>>(s_seq, s_len, s_b);
  dim3 grid(256, 256);
  gemm_scores<<<grid, BLK_T, 0, stream>>>(im_b, s_b, scor);
  loss_kernel<<<1, 256, 0, stream>>>(scor, (float*)d_out);
}
